// Band_49022756717118
// MI455X (gfx1250) — matrix operand order repaired and checked by reading, not compiled, not run on hardware
//
#include <hip/hip_runtime.h>

#define NBANDS 64
#define DLAT   128
#define TT     512
#define FF     1025
#define CC     2
#define BATCH  16
#define BT     (BATCH * TT)          // 8192 rows
#define ROWS_PER_BLOCK 128           // 8 waves x 16 rows

typedef __attribute__((ext_vector_type(16))) __bf16 v16bf;
typedef __attribute__((ext_vector_type(8)))  __bf16 v8bf;
typedef __attribute__((ext_vector_type(2)))  __bf16 v2bf;
typedef __attribute__((ext_vector_type(8)))  float  v8f;

__global__ void zero_out_kernel(float* __restrict__ out, int n) {
  int i = blockIdx.x * blockDim.x + threadIdx.x;
  int stride = gridDim.x * blockDim.x;
  for (; i < n; i += stride) out[i] = 0.0f;
}

__global__ __launch_bounds__(256) void bandsplit_fused(
    const float* __restrict__ x,        // [B,F,T,C]
    const int*   __restrict__ f_idxes,  // [K*W]
    const float* __restrict__ mask,     // [K*W]
    const float* __restrict__ ola,      // [F]
    const float* __restrict__ pre_w,    // [K,WC,128]
    const float* __restrict__ pre_b,    // [K,128]
    const float* __restrict__ post_w,   // [K,128,WC]
    const float* __restrict__ post_b,   // [K,WC]
    float*       __restrict__ out,      // [B,F,T,C]
    int W)
{
  // phase1: preW^T  [o=128][i(pad 64)] ; phase2: postW^T [o(pad 64)][d=128]
  __shared__ __align__(32) __bf16 sW[128 * 64];
  __shared__ __align__(16) __bf16 sH[8][16][136];   // per-wave h tile (bf16)
  __shared__ float sPreB[128];
  __shared__ float sPostB[64];
  __shared__ float sScale[64];   // mask/ola per window slot
  __shared__ float sMask[64];
  __shared__ int   sF[64];

  const int WC   = W * CC;                 // assumed <= 64 (W ~ 30)
  const int band = blockIdx.x & (NBANDS - 1);
  const int rowBlk = blockIdx.x >> 6;
  const int tid  = threadIdx.x;
  const int lane = tid & 31;
  const int wave = tid >> 5;
  const int m    = lane & 15;              // row/col within 16-tile
  const int hi   = lane >> 4;              // lane half

  // ---- phase 1 staging: pre_w transposed + band metadata ----
  for (int idx = tid * 2; idx < 128 * 64; idx += 512) {
    int o = idx >> 6, i = idx & 63;        // i even
    float v0 = (i     < WC) ? pre_w[((size_t)band * WC + i    ) * DLAT + o] : 0.0f;
    float v1 = (i + 1 < WC) ? pre_w[((size_t)band * WC + i + 1) * DLAT + o] : 0.0f;
    *(v2bf*)&sW[o * 64 + i] = (v2bf){(__bf16)v0, (__bf16)v1};
  }
  if (tid < DLAT) sPreB[tid] = pre_b[band * DLAT + tid];
  if (tid < 64) {
    int f = 0; float mv = 0.0f, sc = 0.0f;
    if (tid < W) {
      f  = f_idxes[band * W + tid];
      mv = mask[band * W + tid];
      sc = mv / ola[f];
    }
    sF[tid] = f; sMask[tid] = mv; sScale[tid] = sc;
  }
  __syncthreads();

  // ---- gather A fragments (g tile, 16 x 64, bf16, mask applied) ----
  const int rowBase = rowBlk * ROWS_PER_BLOCK + wave * 16;
  const int r  = rowBase + m;
  const int bb = r >> 9;                   // / T (512)
  const int tt = r & (TT - 1);
  const float* xbt = x + ((size_t)bb * FF * TT + tt) * CC;  // + f*T*C + c

  v16bf afrag[2];
  #pragma unroll
  for (int ks = 0; ks < 2; ++ks) {
    v16bf tmp;
    #pragma unroll
    for (int jp = 0; jp < 8; ++jp) {
      int j = jp * 2;
      // A layout: K = ks*32 + hi*16 + j (contiguous) ; pairs share w
      int kk = ks * 32 + hi * 16 + j;   // even
      float v0 = 0.0f, v1 = 0.0f;
      if (kk < WC) {
        int w = kk >> 1;
        float2 p = *(const float2*)&xbt[(size_t)sF[w] * (TT * CC)];
        float mv = sMask[w];
        v0 = p.x * mv; v1 = p.y * mv;
      }
      tmp[j]     = (__bf16)v0;
      tmp[j + 1] = (__bf16)v1;
    }
    afrag[ks] = tmp;
  }

  // ---- GEMM1: h[16,128] = g[16,64] x preW[64,128] ----
  v8f acc[8] = {};
  #pragma unroll
  for (int ks = 0; ks < 2; ++ks) {
    #pragma unroll
    for (int n = 0; n < 8; ++n) {
      // B layout: lane col = n*16+m, K = ks*32 + hi*16 + j (contiguous)
      v16bf bfrag = *(const v16bf*)&sW[(n * 16 + m) * 64 + ks * 32 + hi * 16];
      acc[n] = __builtin_amdgcn_wmma_f32_16x16x32_bf16(
          false, afrag[ks], false, bfrag, (short)0, acc[n], false, false);
    }
  }

  // ---- h (+bias) -> LDS as bf16, re-layout C-frag -> A-frag ----
  #pragma unroll
  for (int n = 0; n < 8; ++n) {
    float bv = sPreB[n * 16 + m];
    #pragma unroll
    for (int v = 0; v < 8; ++v) {
      // C layout: M = v + 8*hi, N = n*16 + m
      sH[wave][v + 8 * hi][n * 16 + m] = (__bf16)(acc[n][v] + bv);
    }
  }
  __syncthreads();   // all waves done with sW (GEMM1) before overwrite

  // ---- phase 2 staging: post_w transposed [o][d] ----
  for (int idx = tid * 2; idx < 64 * 128; idx += 512) {
    int o = idx >> 7, d = idx & 127;       // d even
    float v0 = 0.0f, v1 = 0.0f;
    if (o < WC) {
      v0 = post_w[((size_t)band * DLAT + d    ) * WC + o];
      v1 = post_w[((size_t)band * DLAT + d + 1) * WC + o];
    }
    *(v2bf*)&sW[o * 128 + d] = (v2bf){(__bf16)v0, (__bf16)v1};
  }
  if (tid < 64) sPostB[tid] = (tid < WC) ? post_b[band * WC + tid] : 0.0f;
  __syncthreads();

  // ---- GEMM2: y[16,64] = h[16,128] x postW[128,64] ----
  v8f acc2[4] = {};
  #pragma unroll
  for (int ks = 0; ks < 4; ++ks) {
    // A-frag of h: row m, K = ks*32 + hi*16 + {0..15}
    const __bf16* hp = &sH[wave][m][ks * 32 + hi * 16];
    v8bf lo = *(const v8bf*)hp;
    v8bf hh = *(const v8bf*)(hp + 8);
    v16bf a2 = __builtin_shufflevector(lo, hh,
        0, 1, 2, 3, 4, 5, 6, 7, 8, 9, 10, 11, 12, 13, 14, 15);
    #pragma unroll
    for (int n2 = 0; n2 < 4; ++n2) {
      v16bf b2 = *(const v16bf*)&sW[(n2 * 16 + m) * 128 + ks * 32 + hi * 16];
      acc2[n2] = __builtin_amdgcn_wmma_f32_16x16x32_bf16(
          false, a2, false, b2, (short)0, acc2[n2], false, false);
    }
  }

  // ---- masked scatter-add with OLA normalization folded in ----
  #pragma unroll
  for (int n2 = 0; n2 < 4; ++n2) {
    int o = n2 * 16 + m;
    if (o < WC) {
      int w = o >> 1, c = o & 1;
      int f = sF[w];
      float sc = sScale[w];       // mask[w] / ola[f]
      float pb = sPostB[o];
      #pragma unroll
      for (int v = 0; v < 8; ++v) {
        int rr = rowBase + v + 8 * hi;
        int b2 = rr >> 9, t2 = rr & (TT - 1);
        size_t addr = (((size_t)b2 * FF + f) * TT + t2) * CC + c;
        atomicAdd(&out[addr], (acc2[n2][v] + pb) * sc);
      }
    }
  }
}

extern "C" void kernel_launch(void* const* d_in, const int* in_sizes, int n_in,
                              void* d_out, int out_size, void* d_ws, size_t ws_size,
                              hipStream_t stream) {
  const float* x       = (const float*)d_in[0];
  const int*   f_idxes = (const int*)  d_in[1];
  const float* mask    = (const float*)d_in[2];
  const float* ola     = (const float*)d_in[3];
  const float* pre_w   = (const float*)d_in[4];
  const float* pre_b   = (const float*)d_in[5];
  const float* post_w  = (const float*)d_in[6];
  const float* post_b  = (const float*)d_in[7];
  float* out = (float*)d_out;

  int W = in_sizes[1] / NBANDS;   // window width per band (~30)

  zero_out_kernel<<<2048, 256, 0, stream>>>(out, out_size);

  int grid = NBANDS * (BT / ROWS_PER_BLOCK);   // 64 * 64 = 4096 blocks
  bandsplit_fused<<<grid, 256, 0, stream>>>(
      x, f_idxes, mask, ola, pre_w, pre_b, post_w, post_b, out, W);
  (void)d_ws; (void)ws_size; (void)n_in;
}
